// Attention_38474317037957
// MI455X (gfx1250) — matrix operand order repaired and checked by reading, not compiled, not run on hardware
//
#include <hip/hip_runtime.h>

// ---------------------------------------------------------------------------
// Problem constants (match reference)
// ---------------------------------------------------------------------------
static constexpr int B_  = 4;
static constexpr int S_  = 2048;
static constexpr int D_  = 2048;
static constexpr int H_  = 16;
static constexpr int KH_ = 4;
static constexpr int HD_ = 128;
static constexpr float SCALE_ = 0.08838834764831845f;  // 1/sqrt(HD)

// ---------------------------------------------------------------------------
// WMMA types (CDNA5 gfx1250, wave32)
// ---------------------------------------------------------------------------
typedef __attribute__((ext_vector_type(16))) __bf16       v16bf;
typedef __attribute__((ext_vector_type(8)))  float        v8f;
typedef __attribute__((ext_vector_type(4)))  unsigned int v4u;

union BfPack { v4u u[2]; v16bf b; };

__device__ __forceinline__ v8f vzero8() {
  v8f z;
#pragma unroll
  for (int e = 0; e < 8; ++e) z[e] = 0.0f;
  return z;
}

// bf16 <-> f32 via bit ops (round-to-nearest-even), storage = unsigned short
__device__ __forceinline__ unsigned short f2bf(float f) {
  unsigned u = __float_as_uint(f);
  return (unsigned short)((u + 0x7FFFu + ((u >> 16) & 1u)) >> 16);
}
__device__ __forceinline__ float bf2f(unsigned short h) {
  return __uint_as_float(((unsigned)h) << 16);
}

// A-operand 16x32 bf16: lane holds K in {koff..koff+7} U {16+koff..16+koff+7}
__device__ __forceinline__ v16bf load_a16(const unsigned short* p_lo,
                                          const unsigned short* p_hi) {
  BfPack t;
  t.u[0] = *(const v4u*)p_lo;
  t.u[1] = *(const v4u*)p_hi;
  return t.b;
}
// B-operand 32x16 bf16: lane holds K in {koff..koff+7} U {16+koff..16+koff+7}
__device__ __forceinline__ v16bf load_b16(const unsigned short* p) {
  BfPack t;
  t.u[0] = *(const v4u*)p;
  t.u[1] = *(const v4u*)(p + 16);
  return t.b;
}

__device__ __forceinline__ v8f wmma_bf16(v16bf a, v16bf b, v8f c) {
  // (neg_a, A, neg_b, B, c_mod, C, reuse_a, reuse_b)
  return __builtin_amdgcn_wmma_f32_16x16x32_bf16(false, a, false, b,
                                                 (short)0, c, false, false);
}

// ---------------------------------------------------------------------------
// Elementwise kernels: converts, transposes, RoPE
// ---------------------------------------------------------------------------
__global__ __launch_bounds__(256) void k_f32_to_bf16(const float* __restrict__ in,
                                                     unsigned short* __restrict__ out,
                                                     int n) {
  int i = blockIdx.x * 256 + threadIdx.x;
  if (i < n) out[i] = f2bf(in[i]);
}

// w is (K x N) row-major f32; wt is (N x K) row-major bf16
__global__ __launch_bounds__(256) void k_convert_transpose(const float* __restrict__ w,
                                                           unsigned short* __restrict__ wt,
                                                           int K, int N) {
  int i = blockIdx.x * 256 + threadIdx.x;
  if (i < K * N) {
    int k = i / N;
    int n = i - k * N;
    wt[(size_t)n * K + k] = f2bf(w[i]);
  }
}

// In-place RoPE on bf16 tensor of shape (B, S, nheads, HD); one thread per pair
__global__ __launch_bounds__(256) void k_rope(unsigned short* __restrict__ t,
                                              const float* __restrict__ cs,
                                              const float* __restrict__ sn,
                                              int nheads) {
  int i = blockIdx.x * 256 + threadIdx.x;
  int total = B_ * S_ * nheads * (HD_ / 2);
  if (i >= total) return;
  int f    = i & (HD_ / 2 - 1);        // HD/2 = 64
  int rest = i >> 6;                   // (b*S + s)*nheads + head
  int s    = (rest / nheads) & (S_ - 1);
  size_t base = (size_t)rest * HD_ + 2 * f;
  float c  = cs[s * (HD_ / 2) + f];
  float sv = sn[s * (HD_ / 2) + f];
  float t0 = bf2f(t[base]);
  float t1 = bf2f(t[base + 1]);
  t[base]     = f2bf(t0 * c - t1 * sv);
  t[base + 1] = f2bf(t0 * sv + t1 * c);
}

// v (B,S,KH,HD) bf16 -> vt (B,KH,HD,S) bf16
__global__ __launch_bounds__(256) void k_transpose_v(const unsigned short* __restrict__ v,
                                                     unsigned short* __restrict__ vt) {
  int i = blockIdx.x * 256 + threadIdx.x;
  int total = B_ * S_ * KH_ * HD_;
  if (i >= total) return;
  int d  = i & (HD_ - 1);
  int kh = (i >> 7) & (KH_ - 1);
  int s  = (i >> 9) & (S_ - 1);
  int b  = i >> 20;                    // S*KH*HD = 2^20 per batch
  vt[(((size_t)(b * KH_ + kh) * HD_) + d) * S_ + s] = v[i];
}

// ---------------------------------------------------------------------------
// WMMA GEMM: C[M,N] = A[M,K] * B[K,N], A bf16 row-major, BT = B^T bf16 (N x K)
// Block: 256 threads = 8 waves arranged 2(M) x 4(N); wave tile 32 x 128
// (2 M-subtiles x 8 N-subtiles = 16 WMMA per 32-deep K step; each B tile is
// register-reused by two WMMAs). Block tile: 64(M) x 512(N).
// ---------------------------------------------------------------------------
template <bool F32OUT>
__global__ __launch_bounds__(256) void k_gemm(const unsigned short* __restrict__ A,
                                              const unsigned short* __restrict__ BT,
                                              void* __restrict__ Cout,
                                              int M, int N, int K) {
  const int wave = threadIdx.x >> 5;
  const int lane = threadIdx.x & 31;
  const int half = lane >> 4;
  const int ln   = lane & 15;
  const int koff = half * 8;
  const int wm = wave & 1, wn = wave >> 1;
  const int m0 = blockIdx.y * 64 + wm * 32;
  const int n0 = blockIdx.x * 512 + wn * 128;
  (void)M;

  v8f acc[2][8];
#pragma unroll
  for (int mi = 0; mi < 2; ++mi)
#pragma unroll
    for (int j = 0; j < 8; ++j) acc[mi][j] = vzero8();

  const unsigned short* arow0 = A + (size_t)(m0 + ln) * K;
  const unsigned short* arow1 = A + (size_t)(m0 + 16 + ln) * K;
  const unsigned short* brow  = BT + (size_t)(n0 + ln) * K;

  for (int kk = 0; kk < K; kk += 32) {
    // speculative prefetch ~8 K-steps ahead for the streamed A operand
    __builtin_prefetch(arow0 + kk + 256, 0, 0);
    __builtin_prefetch(arow1 + kk + 256, 0, 0);

    v16bf a0 = load_a16(arow0 + kk + koff, arow0 + kk + 16 + koff);
    v16bf a1 = load_a16(arow1 + kk + koff, arow1 + kk + 16 + koff);
#pragma unroll
    for (int j = 0; j < 8; ++j) {
      const unsigned short* br = brow + (size_t)(j * 16) * K;
      v16bf b = load_b16(br + kk + half * 8);
      acc[0][j] = wmma_bf16(a0, b, acc[0][j]);
      acc[1][j] = wmma_bf16(a1, b, acc[1][j]);
    }
  }

#pragma unroll
  for (int mi = 0; mi < 2; ++mi) {
#pragma unroll
    for (int j = 0; j < 8; ++j) {
#pragma unroll
      for (int r = 0; r < 8; ++r) {
        int row = m0 + mi * 16 + r + 8 * half;
        int col = n0 + j * 16 + ln;
        if (F32OUT)
          ((float*)Cout)[(size_t)row * N + col] = acc[mi][j][r];
        else
          ((unsigned short*)Cout)[(size_t)row * N + col] = f2bf(acc[mi][j][r]);
      }
    }
  }
}

// ---------------------------------------------------------------------------
// Flash attention (causal, GQA kh = h % KH to match jnp.tile)
// Grid: (S/128, H, B); block 256 = 8 waves; one wave = 16 query rows.
// Key blocks of 32; online softmax; P reshaped C-layout -> A-layout via LDS.
// ---------------------------------------------------------------------------
__global__ __launch_bounds__(256) void k_flash_attn(const unsigned short* __restrict__ q,
                                                    const unsigned short* __restrict__ k,
                                                    const unsigned short* __restrict__ vt,
                                                    unsigned short* __restrict__ o) {
  __shared__ __align__(16) unsigned short plds[8][16 * 32];

  const int wave = threadIdx.x >> 5;
  const int lane = threadIdx.x & 31;
  const int half = lane >> 4;
  const int ln   = lane & 15;
  const int koff = half * 8;

  const int b  = blockIdx.z;
  const int h  = blockIdx.y;
  const int kh = h & (KH_ - 1);            // jnp.tile -> h % KH
  const int q0 = blockIdx.x * 128 + wave * 16;

  // Q tile in A-operand layout: 4 sub-tiles over head dim (128 = 4 * 32)
  v16bf qa[4];
  {
    const unsigned short* qrow =
        q + (((size_t)(b * S_ + q0 + ln) * H_) + h) * HD_;
#pragma unroll
    for (int t = 0; t < 4; ++t)
      qa[t] = load_a16(qrow + t * 32 + koff, qrow + t * 32 + 16 + koff);
  }

  v8f ov[8];
#pragma unroll
  for (int t = 0; t < 8; ++t) ov[t] = vzero8();
  float m_st[8], l_st[8];
#pragma unroll
  for (int r = 0; r < 8; ++r) { m_st[r] = -3.0e38f; l_st[r] = 0.0f; }

  const int kend = q0 + 15;               // last key any row in this tile needs
  for (int kb = 0; kb <= kend; kb += 32) {
    // ---- S = Q K^T for 32 keys (two 16x16 C tiles) ----
    v8f s0 = vzero8(), s1 = vzero8();
    const unsigned short* krow0 =
        k + (((size_t)(b * S_ + kb + ln) * KH_) + kh) * HD_;
    const unsigned short* krow1 =
        k + (((size_t)(b * S_ + kb + 16 + ln) * KH_) + kh) * HD_;
    // prefetch next key block's K rows (speculative; dropped if OOB)
    __builtin_prefetch(krow0 + (size_t)32 * KH_ * HD_, 0, 0);
    __builtin_prefetch(krow1 + (size_t)32 * KH_ * HD_, 0, 0);
#pragma unroll
    for (int t = 0; t < 4; ++t) {
      v16bf b0 = load_b16(krow0 + t * 32 + half * 8);
      v16bf b1 = load_b16(krow1 + t * 32 + half * 8);
      s0 = wmma_bf16(qa[t], b0, s0);
      s1 = wmma_bf16(qa[t], b1, s1);
    }

    // ---- scale (+ causal mask only on the diagonal block) + online softmax --
    const bool needmask = (kb + 31) > q0;  // wave-uniform
    float p0[8], p1[8], alpha[8];
#pragma unroll
    for (int r = 0; r < 8; ++r) {
      int row = q0 + r + 8 * half;
      float x0 = s0[r] * SCALE_;
      float x1 = s1[r] * SCALE_;
      if (needmask) {
        x0 += ((kb + ln)      > row) ? -1.0e9f : 0.0f;
        x1 += ((kb + 16 + ln) > row) ? -1.0e9f : 0.0f;
      }
      float mx = fmaxf(x0, x1);
      mx = fmaxf(mx, __shfl_xor(mx, 1, 32));
      mx = fmaxf(mx, __shfl_xor(mx, 2, 32));
      mx = fmaxf(mx, __shfl_xor(mx, 4, 32));
      mx = fmaxf(mx, __shfl_xor(mx, 8, 32));
      float mnew = fmaxf(m_st[r], mx);
      alpha[r] = __expf(m_st[r] - mnew);
      m_st[r]  = mnew;
      p0[r] = __expf(x0 - mnew);
      p1[r] = __expf(x1 - mnew);
      float rs = p0[r] + p1[r];
      rs += __shfl_xor(rs, 1, 32);
      rs += __shfl_xor(rs, 2, 32);
      rs += __shfl_xor(rs, 4, 32);
      rs += __shfl_xor(rs, 8, 32);
      l_st[r] = l_st[r] * alpha[r] + rs;
    }

    // ---- rescale running output ----
#pragma unroll
    for (int t = 0; t < 8; ++t)
#pragma unroll
      for (int r = 0; r < 8; ++r) ov[t][r] *= alpha[r];

    // ---- P: C-layout -> row-major 16x32 bf16 in LDS -> A-operand ----
#pragma unroll
    for (int r = 0; r < 8; ++r) {
      int row = r + 8 * half;
      plds[wave][row * 32 + ln]      = f2bf(p0[r]);
      plds[wave][row * 32 + 16 + ln] = f2bf(p1[r]);
    }
    // Same-wave LDS ops are in-order; wait for stores before re-reading.
    asm volatile("s_wait_dscnt 0x0" ::: "memory");
    v16bf pa = load_a16(&plds[wave][ln * 32 + koff],
                        &plds[wave][ln * 32 + 16 + koff]);

    // ---- O += P @ V  (vt is (B,KH,HD,S): contiguous along keys) ----
    const unsigned short* vbase =
        vt + ((size_t)(b * KH_ + kh) * HD_) * S_ + kb;
#pragma unroll
    for (int t = 0; t < 8; ++t) {
      const unsigned short* vrow = vbase + (size_t)(t * 16 + ln) * S_;
      __builtin_prefetch(vrow + 32, 0, 0);  // next key block of V
      v16bf vb = load_b16(vrow + half * 8);
      ov[t] = wmma_bf16(pa, vb, ov[t]);
    }
  }

  // ---- finalize: O /= l, store bf16 (B,S,H,HD) ----
#pragma unroll
  for (int r = 0; r < 8; ++r) {
    float inv = 1.0f / l_st[r];
    int row = q0 + r + 8 * half;
    unsigned short* orow = o + (((size_t)(b * S_ + row) * H_) + h) * HD_;
#pragma unroll
    for (int t = 0; t < 8; ++t) orow[t * 16 + ln] = f2bf(ov[t][r] * inv);
  }
}

// ---------------------------------------------------------------------------
// Host launcher
// ---------------------------------------------------------------------------
static inline int ceil_div(long long a, int b) { return (int)((a + b - 1) / b); }

extern "C" void kernel_launch(void* const* d_in, const int* in_sizes, int n_in,
                              void* d_out, int out_size, void* d_ws, size_t ws_size,
                              hipStream_t stream) {
  (void)in_sizes; (void)n_in; (void)out_size; (void)ws_size;

  const float* x    = (const float*)d_in[0];
  const float* wq   = (const float*)d_in[1];
  const float* wk   = (const float*)d_in[2];
  const float* wv   = (const float*)d_in[3];
  const float* wo   = (const float*)d_in[4];
  const float* fcos = (const float*)d_in[5];
  const float* fsin = (const float*)d_in[6];
  // d_in[7] (mask) unused: causal mask applied analytically.

  unsigned short* ws = (unsigned short*)d_ws;
  size_t off = 0;
  auto alloc = [&](size_t elems) { unsigned short* p = ws + off; off += elems; return p; };
  unsigned short* x_bf  = alloc((size_t)B_ * S_ * D_);          // 32 MB
  unsigned short* wqT   = alloc((size_t)D_ * D_);               //  8 MB
  unsigned short* wkT   = alloc((size_t)D_ * KH_ * HD_);        //  2 MB
  unsigned short* wvT   = alloc((size_t)D_ * KH_ * HD_);        //  2 MB
  unsigned short* woT   = alloc((size_t)D_ * D_);               //  8 MB
  unsigned short* q_bf  = alloc((size_t)B_ * S_ * H_ * HD_);    // 32 MB
  unsigned short* k_bf  = alloc((size_t)B_ * S_ * KH_ * HD_);   //  8 MB
  unsigned short* v_bf  = alloc((size_t)B_ * S_ * KH_ * HD_);   //  8 MB
  unsigned short* vT    = alloc((size_t)B_ * KH_ * HD_ * S_);   //  8 MB
  unsigned short* attn  = alloc((size_t)B_ * S_ * H_ * HD_);    // 32 MB

  const int M  = B_ * S_;          // 8192
  const int Nq = H_ * HD_;         // 2048
  const int Nk = KH_ * HD_;        // 512

  // 1) converts
  k_f32_to_bf16<<<ceil_div((long long)M * D_, 256), 256, 0, stream>>>(x, x_bf, M * D_);
  k_convert_transpose<<<ceil_div((long long)D_ * Nq, 256), 256, 0, stream>>>(wq, wqT, D_, Nq);
  k_convert_transpose<<<ceil_div((long long)D_ * Nk, 256), 256, 0, stream>>>(wk, wkT, D_, Nk);
  k_convert_transpose<<<ceil_div((long long)D_ * Nk, 256), 256, 0, stream>>>(wv, wvT, D_, Nk);
  k_convert_transpose<<<ceil_div((long long)Nq * D_, 256), 256, 0, stream>>>(wo, woT, Nq, D_);

  // 2) QKV projections (bf16 WMMA); block tile 64(M) x 512(N)
  k_gemm<false><<<dim3(Nq / 512, M / 64), 256, 0, stream>>>(x_bf, wqT, q_bf, M, Nq, D_);
  k_gemm<false><<<dim3(Nk / 512, M / 64), 256, 0, stream>>>(x_bf, wkT, k_bf, M, Nk, D_);
  k_gemm<false><<<dim3(Nk / 512, M / 64), 256, 0, stream>>>(x_bf, wvT, v_bf, M, Nk, D_);

  // 3) RoPE (in place) on Q and K
  k_rope<<<ceil_div((long long)B_ * S_ * H_ * (HD_ / 2), 256), 256, 0, stream>>>(q_bf, fcos, fsin, H_);
  k_rope<<<ceil_div((long long)B_ * S_ * KH_ * (HD_ / 2), 256), 256, 0, stream>>>(k_bf, fcos, fsin, KH_);

  // 4) V transpose for contiguous-key PV operand
  k_transpose_v<<<ceil_div((long long)B_ * S_ * KH_ * HD_, 256), 256, 0, stream>>>(v_bf, vT);

  // 5) causal flash attention
  k_flash_attn<<<dim3(S_ / 128, H_, B_), 256, 0, stream>>>(q_bf, k_bf, vT, attn);

  // 6) output projection -> fp32 d_out
  k_gemm<true><<<dim3(D_ / 512, M / 64), 256, 0, stream>>>(attn, woT, d_out, M, D_, Nq);
}
